// GraphBlock_1108101563006
// MI455X (gfx1250) — matrix operand order repaired and checked by reading, not compiled, not run on hardware
//
#include <hip/hip_runtime.h>
#include <hip/hip_fp16.h>

#define N_NODES 4096
#define IN_DIM  256
#define OUT_DIM 256
#define HEADS   4
#define HEAD_DIM 64
#define NEG_INF (-9e15f)
#define LN_EPS  1e-5f
#define LOG2E   1.44269504088896340736f

typedef __attribute__((ext_vector_type(16))) _Float16 v16h;
typedef __attribute__((ext_vector_type(8)))  _Float16 v8h;
typedef __attribute__((ext_vector_type(8)))  float    v8f;
typedef __attribute__((ext_vector_type(4)))  unsigned int v4u;
typedef __attribute__((ext_vector_type(8)))  int      v8i;
typedef __attribute__((ext_vector_type(4)))  int      v4i;

__device__ __forceinline__ v8f wmma_f16(v16h a, v16h b, v8f c) {
  // (neg_a, A, neg_b, B, c_mod, C, reuse_a, reuse_b)
  return __builtin_amdgcn_wmma_f32_16x16x32_f16(false, a, false, b, (short)0, c,
                                                false, false);
}

__device__ __forceinline__ void fill8_from_f32(v16h& A, int base, float4 lo, float4 hi) {
  A[base + 0] = (_Float16)lo.x; A[base + 1] = (_Float16)lo.y;
  A[base + 2] = (_Float16)lo.z; A[base + 3] = (_Float16)lo.w;
  A[base + 4] = (_Float16)hi.x; A[base + 5] = (_Float16)hi.y;
  A[base + 6] = (_Float16)hi.z; A[base + 7] = (_Float16)hi.w;
}

__device__ __forceinline__ void fill8_h(v16h& B, int base, v8h v) {
#pragma unroll
  for (int e = 0; e < 8; ++e) B[base + e] = v[e];
}

// Issue a TDM load of a 2D f16 tile (32 cols of j  x  64 rows of c) from a
// row-major [64][4096] f16 tensor slice into LDS (row-major [64][32]).
// D# packed per CDNA5 ISA 8.3/8.4 (group0: count/lds/global/type, group1:
// data_size=2B, tensor_dim0=4096, tensor_dim1=64, tile 32x64, stride0=4096).
// This toolchain exposes the 6-arg builtin: (g0, g1, g2, g3, g4, cpol).
__device__ __forceinline__ void tdm_load_tile(unsigned lds_addr, const _Float16* gsrc) {
  unsigned long long ga = (unsigned long long)gsrc;
  v4u g0;
  g0[0] = 1u;                                    // count=1, is_restore=0, no gather
  g0[1] = lds_addr;                              // lds_addr [63:32]
  g0[2] = (unsigned)ga;                          // global_addr [95:64]
  g0[3] = (unsigned)((ga >> 32) & 0x1FFFFFFull)  // global_addr [120:96]
          | 0x80000000u;                         // type=2 ("image") [127:126]
  v8i g1;
  g1[0] = 0x00010000;                            // workgroup_mask=0, data_size=1 (2B)
  g1[1] = (int)(4096u << 16);                    // tensor_dim0[15:0]=4096 @ [63:48]
  g1[2] = (int)(64u << 16);                      // tensor_dim1[15:0]=64  @ [95:80]
  g1[3] = (int)(32u << 16);                      // tile_dim0=32          @ [127:112]
  g1[4] = 64;                                    // tile_dim1=64 @ [143:128], tile_dim2=0
  g1[5] = 4096;                                  // tensor_dim0_stride[31:0]
  g1[6] = 0;
  g1[7] = 0;
  v4i z4 = {};
  v8i z8 = {};
  __builtin_amdgcn_tensor_load_to_lds(g0, g1, z4, z4, z8, 0);
}

// -------------------- kernel 1: weight prep (transpose + f16) -----------------
// WcatT[c][d] = W[c/64][d][c%64]  (c = head*64+k) ; WrT[c][d] = Wr[d][c]
__global__ void prep_w_kernel(const float* __restrict__ W, const float* __restrict__ Wr,
                              _Float16* __restrict__ WcatT, _Float16* __restrict__ WrT) {
  int c = blockIdx.x;       // 0..255
  int d = threadIdx.x;      // 0..255
  int h = c >> 6, k = c & 63;
  WcatT[c * IN_DIM + d] = (_Float16)W[(h * IN_DIM + d) * HEAD_DIM + k];
  WrT[c * IN_DIM + d]   = (_Float16)Wr[d * OUT_DIM + c];
}

// -------------------- kernel 2: projection GEMMs (WMMA) ----------------------
// hfull = x @ Wcat ; resid = x @ Wr + br.  One wave = one 16x16 tile.
__global__ __launch_bounds__(256) void proj_kernel(
    const float* __restrict__ x, const _Float16* __restrict__ WcatT,
    const _Float16* __restrict__ WrT, const float* __restrict__ br,
    float* __restrict__ hfull, float* __restrict__ resid) {
  int wave = (blockIdx.x * blockDim.x + threadIdx.x) >> 5;  // 0..8191
  int lane = threadIdx.x & 31;
  int which = wave >> 12;      // 0 -> hfull, 1 -> resid
  int t = wave & 4095;
  int r0 = (t >> 4) * 16;      // row tile (256 tiles)
  int c0 = (t & 15) * 16;      // col tile (16 tiles)
  int lr  = lane & 15;
  int sel = lane >> 4;

  const _Float16* WT = which ? WrT : WcatT;
  float* out = which ? resid : hfull;

  const float* xrow = x + (size_t)(r0 + lr) * IN_DIM;
  const _Float16* wrow = WT + (size_t)(c0 + lr) * IN_DIM;

  v8f acc = {};
#pragma unroll 2
  for (int k0 = 0; k0 < IN_DIM; k0 += 32) {
    // A fragment: x rows, f32 -> f16, ISA 16-bit A layout
    int ka = k0 + sel * 8;
    float4 a0 = *(const float4*)(xrow + ka);
    float4 a1 = *(const float4*)(xrow + ka + 4);
    float4 a2 = *(const float4*)(xrow + ka + 16);
    float4 a3 = *(const float4*)(xrow + ka + 20);
    v16h A;
    fill8_from_f32(A, 0, a0, a1);
    fill8_from_f32(A, 8, a2, a3);
    // B fragment: ISA 16-bit layout along K (per-lane column c0+lr)
    int kb = k0 + sel * 8;
    v8h b0 = *(const v8h*)(wrow + kb);
    v8h b1 = *(const v8h*)(wrow + kb + 16);
    v16h B;
    fill8_h(B, 0, b0);
    fill8_h(B, 8, b1);
    acc = wmma_f16(A, B, acc);
  }

  float bias = which ? br[c0 + lr] : 0.0f;
#pragma unroll
  for (int v = 0; v < 8; ++v) {
    int row = r0 + v + sel * 8;
    out[(size_t)row * OUT_DIM + c0 + lr] = acc[v] + bias;
  }
}

// -------------------- kernel 3: hT (f16 transpose of hfull) ------------------
__global__ void hT_kernel(const float* __restrict__ hfull, _Float16* __restrict__ hT) {
  int idx = blockIdx.x * 256 + threadIdx.x;
  int c = idx & 255;
  int n = idx >> 8;
  hT[(size_t)c * N_NODES + n] = (_Float16)hfull[(size_t)n * OUT_DIM + c];
}

// -------------------- kernel 4: s-vectors + per-head max ---------------------
__global__ void svec_kernel(const float* __restrict__ hfull, const float* __restrict__ a1,
                            const float* __restrict__ a2, float* __restrict__ scol,
                            float* __restrict__ srow, unsigned int* __restrict__ smax_enc) {
  int h = blockIdx.y;
  int n = blockIdx.x * 256 + threadIdx.x;
  const float* hv = hfull + (size_t)n * OUT_DIM + h * HEAD_DIM;
  float s1 = 0.f, s2 = 0.f;
#pragma unroll 8
  for (int k = 0; k < HEAD_DIM; ++k) {
    float v = hv[k];
    s1 = fmaf(v, a1[h * HEAD_DIM + k], s1);
    s2 = fmaf(v, a2[h * HEAD_DIM + k], s2);
  }
  scol[h * N_NODES + n] = s1;
  srow[h * N_NODES + n] = s2;
  // wave-level max reduce then one order-preserving atomicMax per wave
  float m = s1;
#pragma unroll
  for (int off = 16; off > 0; off >>= 1) m = fmaxf(m, __shfl_xor(m, off));
  if ((threadIdx.x & 31) == 0) {
    unsigned int b = __float_as_uint(m);
    unsigned int key = (b & 0x80000000u) ? ~b : (b | 0x80000000u);
    atomicMax(smax_enc + h, key);
  }
}

// -------------------- kernel 5: fused masked-softmax attention ----------------
// One block = one head x 128 rows (8 waves x 16-row A tiles). The 32x64 f16
// h-tile (B operand) is DMA'd into double-buffered LDS once per j-step by the
// Tensor Data Mover and shared by all 8 waves.
__device__ __forceinline__ void score4(float* p, int4 msk, float4 s,
                                       float srow_i, float m) {
  float e;
  e = srow_i + s.x; e = fmaxf(e, 0.2f * e); e = (msk.x > 0) ? e : NEG_INF;
  p[0] = exp2f((e - m) * LOG2E);
  e = srow_i + s.y; e = fmaxf(e, 0.2f * e); e = (msk.y > 0) ? e : NEG_INF;
  p[1] = exp2f((e - m) * LOG2E);
  e = srow_i + s.z; e = fmaxf(e, 0.2f * e); e = (msk.z > 0) ? e : NEG_INF;
  p[2] = exp2f((e - m) * LOG2E);
  e = srow_i + s.w; e = fmaxf(e, 0.2f * e); e = (msk.w > 0) ? e : NEG_INF;
  p[3] = exp2f((e - m) * LOG2E);
}

__global__ __launch_bounds__(256) void gat_attn_kernel(
    const int* __restrict__ adj, const _Float16* __restrict__ hT,
    const float* __restrict__ scol, const float* __restrict__ srow,
    const unsigned int* __restrict__ smax_enc, float* __restrict__ hprime) {
  __shared__ __align__(128) _Float16 smem[2][HEAD_DIM][32];  // 8 KB, double buffer

  int head = blockIdx.x & 3;
  int rowgrp = blockIdx.x >> 2;                 // 0..31
  int wv = threadIdx.x >> 5;                    // wave in block 0..7
  int lane = threadIdx.x & 31;
  int i0 = rowgrp * 128 + wv * 16;              // this wave's 16-row tile
  int lr  = lane & 15;
  int sel = lane >> 4;

  // decode per-head global max bound of s_col
  unsigned int key = smax_enc[head];
  unsigned int mb = (key & 0x80000000u) ? (key ^ 0x80000000u) : ~key;
  float smax = __uint_as_float(mb);

  int i = i0 + lr;                              // this lane's output row (A-layout)
  float srow_i = srow[head * N_NODES + i];
  float tmax = srow_i + smax;
  float m = fmaxf(tmax, 0.2f * tmax);           // lrelu monotone -> valid upper bound

  const int* adjrow = adj + (size_t)i * N_NODES;
  const float* sc = scol + head * N_NODES;
  const _Float16* hTh = hT + (size_t)head * HEAD_DIM * N_NODES;

  unsigned lds_base = (unsigned)(unsigned long long)(&smem[0][0][0]);
  bool is_wave0 = (threadIdx.x < 32);

  // prologue: TDM load of first tile into buffer 0
  if (is_wave0) tdm_load_tile(lds_base, hTh /* j0 = 0 */);

  v8f acc[4] = {v8f{}, v8f{}, v8f{}, v8f{}};
  float lsum = 0.0f;

  for (int j0 = 0; j0 < N_NODES; j0 += 32) {
    int cur = (j0 >> 5) & 1;
    if (is_wave0) __builtin_amdgcn_s_wait_tensorcnt(0);  // current tile in LDS
    __syncthreads();                                     // visible to all waves
    if (is_wave0 && (j0 + 32) < N_NODES)                 // prefetch next tile
      tdm_load_tile(lds_base + (cur ^ 1) * 4096, hTh + (j0 + 32));

    // --- P fragment built directly in WMMA A register layout (no LDS) ---
    int ja = j0 + sel * 8;
    __builtin_prefetch(adjrow + ja + 32, 0, 0);          // global_prefetch_b8
    int4 m0 = *(const int4*)(adjrow + ja);
    int4 m1 = *(const int4*)(adjrow + ja + 4);
    int4 m2 = *(const int4*)(adjrow + ja + 16);
    int4 m3 = *(const int4*)(adjrow + ja + 20);
    float4 c0 = *(const float4*)(sc + ja);
    float4 c1 = *(const float4*)(sc + ja + 4);
    float4 c2 = *(const float4*)(sc + ja + 16);
    float4 c3 = *(const float4*)(sc + ja + 20);
    float p[16];
    score4(p + 0,  m0, c0, srow_i, m);
    score4(p + 4,  m1, c1, srow_i, m);
    score4(p + 8,  m2, c2, srow_i, m);
    score4(p + 12, m3, c3, srow_i, m);
    v16h P;
#pragma unroll
    for (int e = 0; e < 16; ++e) { lsum += p[e]; P[e] = (_Float16)p[e]; }

    // --- 4 B fragments from the shared LDS tile (2x ds_load_b128 each) ---
#pragma unroll
    for (int t = 0; t < 4; ++t) {
      const _Float16* hrow = &smem[cur][t * 16 + lr][sel * 8];
      v8h b0 = *(const v8h*)(hrow);
      v8h b1 = *(const v8h*)(hrow + 16);
      v16h B;
      fill8_h(B, 0, b0);
      fill8_h(B, 8, b1);
      acc[t] = wmma_f16(P, B, acc[t]);
    }
  }

  // combine the two half-K partial row sums, then broadcast per C-row inverse
  lsum += __shfl_xor(lsum, 16);
  float inv[8];
#pragma unroll
  for (int v = 0; v < 8; ++v) {
    float sv = __shfl(lsum, v + sel * 8);  // lanes 0..15 hold rows 0..15
    inv[v] = 1.0f / sv;
  }

#pragma unroll
  for (int v = 0; v < 8; ++v) {
    int row = i0 + v + sel * 8;
    float* o = hprime + (size_t)row * OUT_DIM + head * HEAD_DIM + lr;
    o[0]  = acc[0][v] * inv[v];
    o[16] = acc[1][v] * inv[v];
    o[32] = acc[2][v] * inv[v];
    o[48] = acc[3][v] * inv[v];
  }
}

// -------------------- kernel 6: ELU + residual + LayerNorm -------------------
__global__ __launch_bounds__(256) void ln_kernel(
    const float* __restrict__ hprime, const float* __restrict__ resid,
    const float* __restrict__ gamma, const float* __restrict__ beta,
    float* __restrict__ out) {
  int row = blockIdx.x * 8 + (threadIdx.x >> 5);
  int lane = threadIdx.x & 31;
  int c = lane * 8;
  const float* hp = hprime + (size_t)row * OUT_DIM + c;
  const float* rs = resid  + (size_t)row * OUT_DIM + c;

  float4 h0 = *(const float4*)(hp), h1 = *(const float4*)(hp + 4);
  float4 r0 = *(const float4*)(rs), r1 = *(const float4*)(rs + 4);
  float hv[8] = {h0.x, h0.y, h0.z, h0.w, h1.x, h1.y, h1.z, h1.w};
  float rv[8] = {r0.x, r0.y, r0.z, r0.w, r1.x, r1.y, r1.z, r1.w};

  float y[8], s = 0.f, ss = 0.f;
#pragma unroll
  for (int e = 0; e < 8; ++e) {
    float v = hv[e];
    v = (v > 0.f) ? v : (expf(v) - 1.0f);   // ELU
    v += rv[e];
    y[e] = v;
    s += v;
    ss = fmaf(v, v, ss);
  }
#pragma unroll
  for (int off = 16; off > 0; off >>= 1) {
    s  += __shfl_xor(s, off);
    ss += __shfl_xor(ss, off);
  }
  float mean = s * (1.0f / OUT_DIM);
  float var  = ss * (1.0f / OUT_DIM) - mean * mean;
  float rstd = rsqrtf(var + LN_EPS);

  float4 g0 = *(const float4*)(gamma + c), g1 = *(const float4*)(gamma + c + 4);
  float4 b0 = *(const float4*)(beta + c),  b1 = *(const float4*)(beta + c + 4);
  float gv[8] = {g0.x, g0.y, g0.z, g0.w, g1.x, g1.y, g1.z, g1.w};
  float bv[8] = {b0.x, b0.y, b0.z, b0.w, b1.x, b1.y, b1.z, b1.w};

  float o[8];
#pragma unroll
  for (int e = 0; e < 8; ++e) o[e] = fmaf(gv[e], (y[e] - mean) * rstd, bv[e]);
  float* op = out + (size_t)row * OUT_DIM + c;
  *(float4*)(op)     = make_float4(o[0], o[1], o[2], o[3]);
  *(float4*)(op + 4) = make_float4(o[4], o[5], o[6], o[7]);
}

// -------------------- host launch --------------------------------------------
extern "C" void kernel_launch(void* const* d_in, const int* in_sizes, int n_in,
                              void* d_out, int out_size, void* d_ws, size_t ws_size,
                              hipStream_t stream) {
  const float* x     = (const float*)d_in[0];
  const int*   adj   = (const int*)d_in[1];
  const float* W     = (const float*)d_in[2];
  const float* a1    = (const float*)d_in[3];
  const float* a2    = (const float*)d_in[4];
  const float* Wr    = (const float*)d_in[5];
  const float* br    = (const float*)d_in[6];
  const float* gamma = (const float*)d_in[7];
  const float* beta  = (const float*)d_in[8];
  float* out = (float*)d_out;

  char* ws = (char*)d_ws;
  size_t off = 0;
  auto alloc = [&](size_t bytes) -> char* {
    char* p = ws + off;
    off += (bytes + 255) & ~(size_t)255;
    return p;
  };
  float*        hfull  = (float*)alloc((size_t)N_NODES * OUT_DIM * sizeof(float));
  float*        resid  = (float*)alloc((size_t)N_NODES * OUT_DIM * sizeof(float));
  float*        hprime = (float*)alloc((size_t)N_NODES * OUT_DIM * sizeof(float));
  _Float16*     hT     = (_Float16*)alloc((size_t)OUT_DIM * N_NODES * sizeof(_Float16));
  _Float16*     WcatT  = (_Float16*)alloc((size_t)OUT_DIM * IN_DIM * sizeof(_Float16));
  _Float16*     WrT    = (_Float16*)alloc((size_t)OUT_DIM * IN_DIM * sizeof(_Float16));
  float*        scol   = (float*)alloc((size_t)HEADS * N_NODES * sizeof(float));
  float*        srow   = (float*)alloc((size_t)HEADS * N_NODES * sizeof(float));
  unsigned int* smax   = (unsigned int*)alloc(HEADS * sizeof(unsigned int));

  (void)hipMemsetAsync(smax, 0, HEADS * sizeof(unsigned int), stream);

  prep_w_kernel<<<256, 256, 0, stream>>>(W, Wr, WcatT, WrT);
  proj_kernel<<<1024, 256, 0, stream>>>(x, WcatT, WrT, br, hfull, resid);
  hT_kernel<<<4096, 256, 0, stream>>>(hfull, hT);
  svec_kernel<<<dim3(16, 4), 256, 0, stream>>>(hfull, a1, a2, scol, srow, smax);
  gat_attn_kernel<<<128, 256, 0, stream>>>(adj, hT, scol, srow, smax, hprime);
  ln_kernel<<<512, 256, 0, stream>>>(hprime, resid, gamma, beta, out);
}
